// TransformerBlock_42941083025601
// MI455X (gfx1250) — compile-verified, matrix operand order checked by reading, not run on hardware
//
#include <hip/hip_runtime.h>
#include <hip/hip_bf16.h>

// ---------------------------------------------------------------------------
// CDNA5 (gfx1250) transformer block:
//  - all GEMMs + attention via v_wmma_f32_16x16x32_bf16 (wave32 WMMA)
//  - GEMM tiles staged with global_load_async_to_lds_b128 (ASYNCcnt) and
//    double-buffered LDS
// ---------------------------------------------------------------------------

typedef __attribute__((ext_vector_type(16))) __bf16 v16bf;
typedef __attribute__((ext_vector_type(8)))  float  v8f;

#define EMB   1024
#define HEADS 16
#define HDIM  64
#define LN_EPS 1e-5f

// K-index striping for 16-bit A/B fragments (CDNA5 ISA 7.12.2):
// lanes 0-15: VGPR0-3 -> K=0..7, VGPR4-7 -> K=16..23
// lanes 16-31: VGPR0-3 -> K=8..15, VGPR4-7 -> K=24..31
__device__ __forceinline__ int frag_k(int e, int half) {
  int v = e >> 1, p = e & 1;
  return (v < 4) ? (8 * half + 2 * v + p) : (8 + 8 * half + 2 * v + p);
}

__device__ __forceinline__ v8f wmma_bf16(v16bf a, v16bf b, v8f c) {
  return __builtin_amdgcn_wmma_f32_16x16x32_bf16(false, a, false, b,
                                                 (short)0, c, false, false);
}

// Async DMA: 16 bytes per lane, global -> LDS, tracked by ASYNCcnt.
__device__ __forceinline__ void async_copy16(unsigned int lds_addr,
                                             const __bf16* gptr) {
  asm volatile("global_load_async_to_lds_b128 %0, %1, off"
               :: "v"(lds_addr), "v"(gptr)
               : "memory");
}
__device__ __forceinline__ unsigned int lds_lo32(const void* p) {
  return (unsigned int)(unsigned long long)p;  // LDS aperture: offset in [31:0]
}

// ---------------------------------------------------------------------------
// fp32 [K,N] -> bf16 [N,K] transposed cast (32x32 tiles through LDS).
__global__ void cast_transpose_bf16(const float* __restrict__ in,
                                    __bf16* __restrict__ out, int K, int N) {
  __shared__ float t[32][33];
  const int k0 = blockIdx.y * 32, n0 = blockIdx.x * 32;
  const int tx = threadIdx.x & 31, ty = threadIdx.x >> 5;  // 32 x 8
#pragma unroll
  for (int i = 0; i < 32; i += 8)
    t[ty + i][tx] = in[(size_t)(k0 + ty + i) * N + n0 + tx];
  __syncthreads();
#pragma unroll
  for (int i = 0; i < 32; i += 8)
    out[(size_t)(n0 + ty + i) * K + k0 + tx] = (__bf16)t[tx][ty + i];
}

// One 256-thread block per row of D=1024.
__global__ void layernorm_bf16(const float* __restrict__ x,
                               const float* __restrict__ g,
                               const float* __restrict__ s,
                               __bf16* __restrict__ out) {
  const int row = blockIdx.x;
  const float* xr = x + (size_t)row * EMB;
  __shared__ float red[16];
  float v[4];
  float sum = 0.f, sq = 0.f;
#pragma unroll
  for (int i = 0; i < 4; ++i) {
    v[i] = xr[threadIdx.x + i * 256];
    sum += v[i];
    sq  += v[i] * v[i];
  }
#pragma unroll
  for (int off = 16; off; off >>= 1) {
    sum += __shfl_down(sum, off, 32);
    sq  += __shfl_down(sq,  off, 32);
  }
  if ((threadIdx.x & 31) == 0) {
    red[threadIdx.x >> 5]       = sum;
    red[8 + (threadIdx.x >> 5)] = sq;
  }
  __syncthreads();
  if (threadIdx.x == 0) {
    float ts = 0.f, tq = 0.f;
#pragma unroll
    for (int i = 0; i < 8; ++i) { ts += red[i]; tq += red[8 + i]; }
    float mean = ts * (1.0f / EMB);
    float var  = tq * (1.0f / EMB) - mean * mean;
    red[0] = mean;
    red[1] = rsqrtf(var + LN_EPS);
  }
  __syncthreads();
  float mean = red[0], inv = red[1];
#pragma unroll
  for (int i = 0; i < 4; ++i) {
    int c = threadIdx.x + i * 256;
    out[(size_t)row * EMB + c] = (__bf16)(g[c] * ((v[i] - mean) * inv) + s[c]);
  }
}

// ---------------------------------------------------------------------------
// Tiled bf16 WMMA GEMM: C[M,N] = act(A[M,K] @ B[K,N] + bias) + resid
// B supplied pre-transposed as BT[N,K] so both tiles are K-contiguous and
// can be staged verbatim with async b128 copies (double-buffered LDS).
// Block tile 64x64, K-step 32; 8 waves, each owns a 16(M)x32(N) sub-tile.
template <bool RELU, bool OUTBF>
__global__ void gemm_bf16_wmma(const __bf16* __restrict__ A,   // [M,K]
                               const __bf16* __restrict__ BT,  // [N,K]
                               const float* __restrict__ bias,
                               const float* __restrict__ resid,
                               void* __restrict__ outp,
                               int M, int N, int K) {
  __shared__ __align__(16) __bf16 As[2][64 * 32];
  __shared__ __align__(16) __bf16 Bs[2][64 * 32];

  const int tid    = threadIdx.x;
  const int lane   = tid & 31;
  const int wave   = tid >> 5;
  const int half   = lane >> 4;
  const int ln     = lane & 15;
  const int mstrip = wave & 3;   // 4 M-strips of 16 rows
  const int ngrp   = wave >> 2;  // 2 N-groups of 32 cols
  const int m0 = blockIdx.y * 64;
  const int n0 = blockIdx.x * 64;

  // Staging: each thread copies one 16B chunk per tile (64 rows x 32 K).
  const int srow = tid >> 2, scol = (tid & 3) * 8;
  const __bf16* aSrc = A  + (size_t)(m0 + srow) * K + scol;
  const __bf16* bSrc = BT + (size_t)(n0 + srow) * K + scol;
  const unsigned int ldsA0 = lds_lo32(&As[0][srow * 32 + scol]);
  const unsigned int ldsA1 = lds_lo32(&As[1][srow * 32 + scol]);
  const unsigned int ldsB0 = lds_lo32(&Bs[0][srow * 32 + scol]);
  const unsigned int ldsB1 = lds_lo32(&Bs[1][srow * 32 + scol]);

  v8f acc0 = {}, acc1 = {};

  async_copy16(ldsA0, aSrc);
  async_copy16(ldsB0, bSrc);

  int buf = 0;
  for (int k0 = 0; k0 < K; k0 += 32) {
    if (k0 + 32 < K) {
      async_copy16(buf ? ldsA0 : ldsA1, aSrc + k0 + 32);
      async_copy16(buf ? ldsB0 : ldsB1, bSrc + k0 + 32);
      asm volatile("s_wait_asynccnt 0x2" ::: "memory");  // current tile landed
    } else {
      asm volatile("s_wait_asynccnt 0x0" ::: "memory");
    }
    __syncthreads();  // all waves' DMA visible

    const __bf16* a  = As[buf];
    const __bf16* bs = Bs[buf];
    v16bf af, b0, b1;
#pragma unroll
    for (int e = 0; e < 16; ++e) {
      int kk = frag_k(e, half);
      af[e] = a [(mstrip * 16 + ln) * 32 + kk];
      b0[e] = bs[(ngrp * 32 + ln) * 32 + kk];
      b1[e] = bs[(ngrp * 32 + 16 + ln) * 32 + kk];
    }
    acc0 = wmma_bf16(af, b0, acc0);
    acc1 = wmma_bf16(af, b1, acc1);
    __syncthreads();  // protect buf^1 before next prefetch overwrites it
    buf ^= 1;
  }

#pragma unroll
  for (int r = 0; r < 8; ++r) {
    int row = m0 + mstrip * 16 + half * 8 + r;
    int c0  = n0 + ngrp * 32 + ln;
    int c1  = c0 + 16;
    float v0 = acc0[r], v1 = acc1[r];
    if (bias) { v0 += bias[c0]; v1 += bias[c1]; }
    if (RELU) { v0 = fmaxf(v0, 0.f); v1 = fmaxf(v1, 0.f); }
    if (resid) {
      v0 += resid[(size_t)row * N + c0];
      v1 += resid[(size_t)row * N + c1];
    }
    if (OUTBF) {
      __bf16* o = (__bf16*)outp;
      o[(size_t)row * N + c0] = (__bf16)v0;
      o[(size_t)row * N + c1] = (__bf16)v1;
    } else {
      float* o = (float*)outp;
      o[(size_t)row * N + c0] = v0;
      o[(size_t)row * N + c1] = v1;
    }
  }
}

// ---------------------------------------------------------------------------
// Causal flash attention, one wave per (b, h, 16-query tile).
// Q/K/V/O are bf16 [B*T, EMB] with head h at column h*64.
__global__ void flash_attn_wmma(const __bf16* __restrict__ Q,
                                const __bf16* __restrict__ Km,
                                const __bf16* __restrict__ V,
                                __bf16* __restrict__ O, int T) {
  const int q0   = blockIdx.x * 16;
  const int h    = blockIdx.y;
  const int b    = blockIdx.z;
  const int lane = threadIdx.x & 31;
  const int half = lane >> 4;
  const int ln   = lane & 15;
  const size_t baseRow = (size_t)b * T;
  const int hoff = h * HDIM;

  __shared__ __bf16 Pl[16 * 32];

  // Q A-fragments: two K=32 chunks across head_dim 64
  v16bf qa[2];
#pragma unroll
  for (int c = 0; c < 2; ++c)
#pragma unroll
    for (int e = 0; e < 16; ++e)
      qa[c][e] = Q[(baseRow + q0 + ln) * EMB + hoff + c * 32 + frag_k(e, half)];

  float mi[8], li[8];
  v8f o[4] = {};
#pragma unroll
  for (int r = 0; r < 8; ++r) { mi[r] = -3.0e38f; li[r] = 0.f; }

  const float scale = 0.03125f;  // 1/sqrt(EMB) per reference
  const int nkt = (q0 + 16 + 31) / 32;
  for (int kt = 0; kt < nkt; ++kt) {
    const int kb = kt * 32;

    // S = Q @ K^T for 32 keys (two 16x16 tiles)
    v8f s0 = {}, s1 = {};
#pragma unroll
    for (int c = 0; c < 2; ++c) {
      v16bf kb0, kb1;
#pragma unroll
      for (int e = 0; e < 16; ++e) {
        int kk = frag_k(e, half);
        kb0[e] = Km[(baseRow + kb + ln) * EMB + hoff + c * 32 + kk];
        kb1[e] = Km[(baseRow + kb + 16 + ln) * EMB + hoff + c * 32 + kk];
      }
      s0 = wmma_bf16(qa[c], kb0, s0);
      s1 = wmma_bf16(qa[c], kb1, s1);
    }

    // online softmax (rows live in one 16-lane half at register r)
    float p0[8], p1[8];
#pragma unroll
    for (int r = 0; r < 8; ++r) {
      int qrow = q0 + half * 8 + r;
      float v0 = (kb + ln      <= qrow) ? s0[r] * scale : -3.0e38f;
      float v1 = (kb + 16 + ln <= qrow) ? s1[r] * scale : -3.0e38f;
      float rmax = fmaxf(v0, v1);
#pragma unroll
      for (int off = 1; off < 16; off <<= 1)
        rmax = fmaxf(rmax, __shfl_xor(rmax, off, 32));
      float mnew  = fmaxf(mi[r], rmax);
      float alpha = __expf(mi[r] - mnew);
      v0 = __expf(v0 - mnew);
      v1 = __expf(v1 - mnew);
      float rsum = v0 + v1;
#pragma unroll
      for (int off = 1; off < 16; off <<= 1)
        rsum += __shfl_xor(rsum, off, 32);
      li[r] = li[r] * alpha + rsum;
      mi[r] = mnew;
#pragma unroll
      for (int dd = 0; dd < 4; ++dd) o[dd][r] *= alpha;
      p0[r] = v0;
      p1[r] = v1;
    }

    // stage P through LDS to re-layout C-frag -> A-frag (16x32)
#pragma unroll
    for (int r = 0; r < 8; ++r) {
      int m = half * 8 + r;
      Pl[m * 32 + ln]      = (__bf16)p0[r];
      Pl[m * 32 + 16 + ln] = (__bf16)p1[r];
    }
    __syncthreads();
    v16bf pa;
#pragma unroll
    for (int e = 0; e < 16; ++e) pa[e] = Pl[ln * 32 + frag_k(e, half)];

    // O += P @ V  (four 16-wide d-chunks)
#pragma unroll
    for (int dd = 0; dd < 4; ++dd) {
      v16bf vb;
#pragma unroll
      for (int e = 0; e < 16; ++e) {
        int kk = frag_k(e, half);
        vb[e] = V[(baseRow + kb + kk) * EMB + hoff + dd * 16 + ln];
      }
      o[dd] = wmma_bf16(pa, vb, o[dd]);
    }
    __syncthreads();
  }

#pragma unroll
  for (int r = 0; r < 8; ++r) {
    float inv = 1.0f / li[r];
    int qrow = q0 + half * 8 + r;
#pragma unroll
    for (int dd = 0; dd < 4; ++dd)
      O[(baseRow + qrow) * EMB + hoff + dd * 16 + ln] = (__bf16)(o[dd][r] * inv);
  }
}

// ---------------------------------------------------------------------------
extern "C" void kernel_launch(void* const* d_in, const int* in_sizes, int n_in,
                              void* d_out, int out_size, void* d_ws,
                              size_t ws_size, hipStream_t stream) {
  (void)in_sizes; (void)n_in; (void)out_size; (void)ws_size;
  const int B = 2, T = 2048, D = EMB, M = B * T, D4 = 4 * D;

  const float* x  = (const float*)d_in[0];
  const float* Wq = (const float*)d_in[1];
  const float* Wk = (const float*)d_in[2];
  const float* Wv = (const float*)d_in[3];
  const float* Wo = (const float*)d_in[4];
  const float* bo = (const float*)d_in[5];
  const float* W1 = (const float*)d_in[6];
  const float* b1 = (const float*)d_in[7];
  const float* W2 = (const float*)d_in[8];
  const float* b2 = (const float*)d_in[9];
  const float* g1 = (const float*)d_in[10];
  const float* s1 = (const float*)d_in[11];
  const float* g2 = (const float*)d_in[12];
  const float* s2 = (const float*)d_in[13];
  float* out = (float*)d_out;

  size_t off = 0;
  auto alloc = [&](size_t bytes) {
    void* p = (char*)d_ws + off;
    off += (bytes + 255) & ~(size_t)255;
    return p;
  };
  __bf16* Wq_b = (__bf16*)alloc((size_t)D * D * 2);   // [N,K] transposed
  __bf16* Wk_b = (__bf16*)alloc((size_t)D * D * 2);
  __bf16* Wv_b = (__bf16*)alloc((size_t)D * D * 2);
  __bf16* Wo_b = (__bf16*)alloc((size_t)D * D * 2);
  __bf16* W1_b = (__bf16*)alloc((size_t)D * D4 * 2);
  __bf16* W2_b = (__bf16*)alloc((size_t)D4 * D * 2);
  __bf16* h1   = (__bf16*)alloc((size_t)M * D * 2);
  __bf16* q    = (__bf16*)alloc((size_t)M * D * 2);
  __bf16* k    = (__bf16*)alloc((size_t)M * D * 2);
  __bf16* v    = (__bf16*)alloc((size_t)M * D * 2);
  __bf16* attn = (__bf16*)alloc((size_t)M * D * 2);
  float*  xmid = (float*) alloc((size_t)M * D * 4);
  __bf16* h2   = (__bf16*)alloc((size_t)M * D * 2);
  __bf16* ffn  = (__bf16*)alloc((size_t)M * D4 * 2);

  // one-time fp32 [K,N] -> bf16 [N,K] weight transposes
  auto castT = [&](const float* src, __bf16* dst, int Kd, int Nd) {
    cast_transpose_bf16<<<dim3(Nd / 32, Kd / 32), 256, 0, stream>>>(src, dst, Kd, Nd);
  };
  castT(Wq, Wq_b, D, D);
  castT(Wk, Wk_b, D, D);
  castT(Wv, Wv_b, D, D);
  castT(Wo, Wo_b, D, D);
  castT(W1, W1_b, D, D4);
  castT(W2, W2_b, D4, D);

  // LN1 -> h1
  layernorm_bf16<<<M, 256, 0, stream>>>(x, g1, s1, h1);

  // QKV projections
  dim3 gQKV(D / 64, M / 64);
  gemm_bf16_wmma<false, true><<<gQKV, 256, 0, stream>>>(h1, Wq_b, nullptr, nullptr, q, M, D, D);
  gemm_bf16_wmma<false, true><<<gQKV, 256, 0, stream>>>(h1, Wk_b, nullptr, nullptr, k, M, D, D);
  gemm_bf16_wmma<false, true><<<gQKV, 256, 0, stream>>>(h1, Wv_b, nullptr, nullptr, v, M, D, D);

  // causal flash attention
  flash_attn_wmma<<<dim3(T / 16, HEADS, B), 32, 0, stream>>>(q, k, v, attn, T);

  // out projection + bias + residual(x) -> xmid (f32)
  gemm_bf16_wmma<false, false><<<gQKV, 256, 0, stream>>>(attn, Wo_b, bo, x, xmid, M, D, D);

  // LN2 -> h2
  layernorm_bf16<<<M, 256, 0, stream>>>(xmid, g2, s2, h2);

  // FFN
  gemm_bf16_wmma<true, true><<<dim3(D4 / 64, M / 64), 256, 0, stream>>>(h2, W1_b, b1, nullptr, ffn, M, D4, D);
  gemm_bf16_wmma<false, false><<<dim3(D / 64, M / 64), 256, 0, stream>>>(ffn, W2_b, b2, xmid, out, M, D, D4);
}
